// GPT2Model_89661737271715
// MI455X (gfx1250) — matrix operand order repaired and checked by reading, not compiled, not run on hardware
//
#include <hip/hip_runtime.h>
#include <hip/hip_bf16.h>

typedef __attribute__((ext_vector_type(16))) _Float16 v16h;
typedef __attribute__((ext_vector_type(8)))  float    v8f;

// GPT-2 small dims
#define GC   768
#define GH   12
#define GL   12
#define GB   2
#define GT   1024
#define GD   64
#define GV   50257
#define GVP  50304          // V padded to multiple of 64
#define GBT  (GB*GT)        // 2048 tokens

// ---------------------------------------------------------------------------
// Packed WMMA fragment layouts (CDNA5 ISA 7.12.2, wave32)
//   A (16x32 f16): lane = (m%16) + 16*g,  K = (h<8?h:h+8) + 8*g
//   B (32x16 f16): lane = (n%16) + 16*g,  K = (h<8?h:h+8) + 8*g
// Tiles stored tile-major: ((tile_outer*Ktiles + kt)*32 + lane)*16 + h
// ---------------------------------------------------------------------------
__device__ __forceinline__ size_t a_off(int m, int k, int Ktiles) {
    int kk   = k & 31;
    int h    = (kk & 7) + ((kk >> 4) << 3);
    int lane = (m & 15) + (((kk >> 3) & 1) << 4);
    return ((((size_t)(m >> 4)) * Ktiles + (k >> 5)) * 32 + lane) * 16 + h;
}
__device__ __forceinline__ size_t b_off(int n, int k, int Ktiles) {
    int kk   = k & 31;
    int lane = (n & 15) + (((kk >> 3) & 1) << 4);
    int h    = (kk & 7) + ((kk >> 4) << 3);
    return ((((size_t)(n >> 4)) * Ktiles + (k >> 5)) * 32 + lane) * 16 + h;
}

// ---------------------------------------------------------------------------
// Elementwise / pack kernels
// ---------------------------------------------------------------------------
__global__ void embed_kernel(const int* __restrict__ idx,
                             const float* __restrict__ wte,
                             const float* __restrict__ wpe,
                             float* __restrict__ x) {
    size_t i = (size_t)blockIdx.x * blockDim.x + threadIdx.x;
    if (i >= (size_t)GBT * GC) return;
    int c   = (int)(i % GC);
    int row = (int)(i / GC);
    int t   = row & (GT - 1);
    int tok = idx[row];
    x[i] = wte[(size_t)tok * GC + c] + wpe[(size_t)t * GC + c];
}

// fp32 [K,N] row-major weight -> packed-B f16 (N padded with zeros to Npad)
__global__ void pack_B_kernel(const float* __restrict__ W,
                              _Float16* __restrict__ Bp,
                              int K, int N, int Npad) {
    size_t i = (size_t)blockIdx.x * blockDim.x + threadIdx.x;
    size_t total = (size_t)K * Npad;
    if (i >= total) return;
    int n = (int)(i % Npad);
    int k = (int)(i / Npad);
    float v = (n < N) ? W[(size_t)k * N + n] : 0.0f;
    Bp[b_off(n, k, K >> 5)] = (_Float16)v;
}

// fp32 [M,K] row-major activation -> packed-A f16
__global__ void pack_A_kernel(const float* __restrict__ X,
                              _Float16* __restrict__ Ap,
                              int K, size_t total) {
    size_t i = (size_t)blockIdx.x * blockDim.x + threadIdx.x;
    if (i >= total) return;
    int k = (int)(i % K);
    int m = (int)(i / K);
    Ap[a_off(m, k, K >> 5)] = (_Float16)X[i];
}

// LayerNorm over C=768 per row, write packed-A f16 directly
__global__ void ln_pack_kernel(const float* __restrict__ x,
                               const float* __restrict__ w,
                               const float* __restrict__ b,
                               _Float16* __restrict__ Ap) {
    __shared__ float red[256];
    const int row = blockIdx.x;
    const float* xr = x + (size_t)row * GC;
    const int tid = threadIdx.x;

    float s = 0.0f;
    for (int c = tid; c < GC; c += 256) s += xr[c];
    red[tid] = s; __syncthreads();
    for (int o = 128; o > 0; o >>= 1) { if (tid < o) red[tid] += red[tid + o]; __syncthreads(); }
    const float mu = red[0] * (1.0f / GC);
    __syncthreads();

    float s2 = 0.0f;
    for (int c = tid; c < GC; c += 256) { float d = xr[c] - mu; s2 += d * d; }
    red[tid] = s2; __syncthreads();
    for (int o = 128; o > 0; o >>= 1) { if (tid < o) red[tid] += red[tid + o]; __syncthreads(); }
    const float rstd = rsqrtf(red[0] * (1.0f / GC) + 1e-5f);
    __syncthreads();

    const int Kt = GC >> 5;
    for (int c = tid; c < GC; c += 256) {
        float v = (xr[c] - mu) * rstd * w[c] + b[c];
        Ap[a_off(row, c, Kt)] = (_Float16)v;
    }
}

// split qkv fp32 [2048, 2304] into packed q(A, scaled), k(B over d), v(B over t)
__global__ void qkv_pack_kernel(const float* __restrict__ qkv,
                                _Float16* __restrict__ qA,
                                _Float16* __restrict__ kB,
                                _Float16* __restrict__ vB) {
    size_t i = (size_t)blockIdx.x * blockDim.x + threadIdx.x;
    const size_t total = (size_t)GBT * 3 * GC;
    if (i >= total) return;
    int col = (int)(i % (3 * GC));
    int row = (int)(i / (3 * GC));
    int b   = row >> 10;
    int t   = row & (GT - 1);
    int sec = col / GC;
    int cc  = col - sec * GC;
    int h   = cc >> 6;
    int d   = cc & 63;
    size_t bh = (size_t)(b * GH + h);
    const size_t bhStride = (size_t)GT * GD;   // 65536
    float v = qkv[i];
    if (sec == 0)      qA[bh * bhStride + a_off(t, d, 2)]  = (_Float16)(v * 0.125f); // 1/sqrt(64)
    else if (sec == 1) kB[bh * bhStride + b_off(t, d, 2)]  = (_Float16)v;  // n=t_key, k=d
    else               vB[bh * bhStride + b_off(d, t, 32)] = (_Float16)v;  // n=d,     k=t_key
}

// row softmax over 1024 keys, write probs as packed-A f16 (Ktiles=32)
__global__ void softmax_pack_kernel(const float* __restrict__ scores,
                                    _Float16* __restrict__ Pp) {
    __shared__ float red[256];
    const int z  = blockIdx.x;            // bh*1024 + tq
    const int bh = z >> 10;
    const int tq = z & (GT - 1);
    const float* sr = scores + (size_t)z * GT;
    const int tid = threadIdx.x;

    float mx = -3.0e38f;
    for (int c = tid; c < GT; c += 256) mx = fmaxf(mx, sr[c]);
    red[tid] = mx; __syncthreads();
    for (int o = 128; o > 0; o >>= 1) { if (tid < o) red[tid] = fmaxf(red[tid], red[tid + o]); __syncthreads(); }
    mx = red[0]; __syncthreads();

    float s = 0.0f;
    for (int c = tid; c < GT; c += 256) s += __expf(sr[c] - mx);
    red[tid] = s; __syncthreads();
    for (int o = 128; o > 0; o >>= 1) { if (tid < o) red[tid] += red[tid + o]; __syncthreads(); }
    const float inv = 1.0f / red[0];
    __syncthreads();

    _Float16* P = Pp + (size_t)bh * ((size_t)GT * GT);
    for (int c = tid; c < GT; c += 256)
        P[a_off(tq, c, 32)] = (_Float16)(__expf(sr[c] - mx) * inv);
}

// ---------------------------------------------------------------------------
// Generic WMMA GEMM: one wave -> 64x64 fp32 tile (16 wmma / 8 frag-loads per
// k-step; ~190 VGPRs, fine for wave32's 1024-VGPR budget).
//   mode 0: out = acc + bias
//   mode 1: out = gelu(acc + bias)      (exact erf)
//   mode 2: out = out + acc + bias      (residual)
// Batched over blockIdx.z; output base = (z/HH)*oSO + (z%HH)*oSI.
// ---------------------------------------------------------------------------
__global__ void __launch_bounds__(32)
wmma_gemm_kernel(const _Float16* __restrict__ Apack,
                 const _Float16* __restrict__ Bpack,
                 const float* __restrict__ bias,
                 float* __restrict__ out,
                 int Ktiles, int Nvalid, int ldOut,
                 long long aStride, long long bStride,
                 int HHp, long long oSO, long long oSI,
                 int mode) {
    const int lane = threadIdx.x;
    const int z = blockIdx.z;
    const _Float16* A  = Apack + (size_t)z * aStride;
    const _Float16* Bp = Bpack + (size_t)z * bStride;
    float* O = out + (size_t)(z / HHp) * oSO + (size_t)(z % HHp) * oSI;

    const int mt0 = blockIdx.y * 4;   // four 16-row tiles
    const int nt0 = blockIdx.x * 4;   // four 16-col tiles

    v8f acc[4][4] = {};
    const _Float16* a0p = A  + ((size_t)mt0 * Ktiles) * 512 + lane * 16;
    const _Float16* b0p = Bp + ((size_t)nt0 * Ktiles) * 512 + lane * 16;
    const size_t tS = (size_t)Ktiles * 512;   // elems per 16-row/col tile strip

    for (int kt = 0; kt < Ktiles; ++kt) {
        v16h a0 = *(const v16h*)(a0p + (size_t)kt * 512);
        v16h a1 = *(const v16h*)(a0p + tS     + (size_t)kt * 512);
        v16h a2 = *(const v16h*)(a0p + 2 * tS + (size_t)kt * 512);
        v16h a3 = *(const v16h*)(a0p + 3 * tS + (size_t)kt * 512);
#pragma unroll
        for (int ni = 0; ni < 4; ++ni) {
            v16h bf = *(const v16h*)(b0p + (size_t)ni * tS + (size_t)kt * 512);
            acc[0][ni] = __builtin_amdgcn_wmma_f32_16x16x32_f16(
                false, a0, false, bf, (short)0, acc[0][ni], false, false);
            acc[1][ni] = __builtin_amdgcn_wmma_f32_16x16x32_f16(
                false, a1, false, bf, (short)0, acc[1][ni], false, false);
            acc[2][ni] = __builtin_amdgcn_wmma_f32_16x16x32_f16(
                false, a2, false, bf, (short)0, acc[2][ni], false, false);
            acc[3][ni] = __builtin_amdgcn_wmma_f32_16x16x32_f16(
                false, a3, false, bf, (short)0, acc[3][ni], false, false);
        }
    }

    // D layout: n = lane%16; m = r + 8*(lane/16)
    const int nLane = lane & 15;
    const int mBase = (lane >> 4) * 8;
#pragma unroll
    for (int mi = 0; mi < 4; ++mi) {
#pragma unroll
        for (int ni = 0; ni < 4; ++ni) {
            int n = (nt0 + ni) * 16 + nLane;
            if (n >= Nvalid) continue;
            float bv = bias ? bias[n] : 0.0f;
#pragma unroll
            for (int r = 0; r < 8; ++r) {
                int m = (mt0 + mi) * 16 + mBase + r;
                float v = acc[mi][ni][r] + bv;
                size_t o = (size_t)m * ldOut + n;
                if (mode == 1)      v = 0.5f * v * (1.0f + erff(v * 0.70710678118654752f));
                else if (mode == 2) v += O[o];
                O[o] = v;
            }
        }
    }
}

// ---------------------------------------------------------------------------
// Host driver
// ---------------------------------------------------------------------------
static inline void* ws_take(char*& p, size_t bytes) {
    void* r = (void*)p;
    p += (bytes + 255) & ~(size_t)255;
    return r;
}

extern "C" void kernel_launch(void* const* d_in, const int* in_sizes, int n_in,
                              void* d_out, int out_size, void* d_ws, size_t ws_size,
                              hipStream_t stream) {
    (void)in_sizes; (void)n_in; (void)out_size; (void)ws_size;

    const int*   idx     = (const int*)  d_in[0];
    const float* wte     = (const float*)d_in[1];
    const float* wpe     = (const float*)d_in[2];
    const float* ln1_w   = (const float*)d_in[3];
    const float* ln1_b   = (const float*)d_in[4];
    const float* qkv_w   = (const float*)d_in[5];
    const float* qkv_b   = (const float*)d_in[6];
    const float* proj_w  = (const float*)d_in[7];
    const float* proj_b  = (const float*)d_in[8];
    const float* ln2_w   = (const float*)d_in[9];
    const float* ln2_b   = (const float*)d_in[10];
    const float* fc_w    = (const float*)d_in[11];
    const float* fc_b    = (const float*)d_in[12];
    const float* fc2_w   = (const float*)d_in[13];
    const float* fc2_b   = (const float*)d_in[14];
    const float* lnf_w   = (const float*)d_in[15];
    const float* lnf_b   = (const float*)d_in[16];
    const float* lm_w    = (const float*)d_in[17];
    float* logits = (float*)d_out;

    char* p = (char*)d_ws;
    // packed f16 weights
    _Float16* qkvB = (_Float16*)ws_take(p, (size_t)GL * GC * (3*GC) * 2);
    _Float16* prjB = (_Float16*)ws_take(p, (size_t)GL * GC * GC * 2);
    _Float16* fcB  = (_Float16*)ws_take(p, (size_t)GL * GC * (4*GC) * 2);
    _Float16* fc2B = (_Float16*)ws_take(p, (size_t)GL * (4*GC) * GC * 2);
    _Float16* lmB  = (_Float16*)ws_take(p, (size_t)GC * GVP * 2);
    // activations
    float*    x    = (float*)   ws_take(p, (size_t)GBT * GC * 4);
    _Float16* hA   = (_Float16*)ws_take(p, (size_t)GBT * GC * 2);
    float*    qkv  = (float*)   ws_take(p, (size_t)GBT * (3*GC) * 4);
    _Float16* qA   = (_Float16*)ws_take(p, (size_t)GB * GH * GT * GD * 2);
    _Float16* kB   = (_Float16*)ws_take(p, (size_t)GB * GH * GT * GD * 2);
    _Float16* vB   = (_Float16*)ws_take(p, (size_t)GB * GH * GT * GD * 2);
    float*    sc   = (float*)   ws_take(p, (size_t)GB * GH * GT * GT * 4);
    _Float16* Pp   = (_Float16*)ws_take(p, (size_t)GB * GH * GT * GT * 2);
    float*    y    = (float*)   ws_take(p, (size_t)GBT * GC * 4);
    _Float16* yA   = (_Float16*)ws_take(p, (size_t)GBT * GC * 2);
    float*    h2   = (float*)   ws_take(p, (size_t)GBT * (4*GC) * 4);
    _Float16* h2A  = (_Float16*)ws_take(p, (size_t)GBT * (4*GC) * 2);

    auto packB = [&](const float* W, _Float16* Bp, int K, int N, int Npad) {
        size_t total = (size_t)K * Npad;
        unsigned blocks = (unsigned)((total + 255) / 256);
        pack_B_kernel<<<blocks, 256, 0, stream>>>(W, Bp, K, N, Npad);
    };
    auto packA = [&](const float* X, _Float16* Ap, int M, int K) {
        size_t total = (size_t)M * K;
        unsigned blocks = (unsigned)((total + 255) / 256);
        pack_A_kernel<<<blocks, 256, 0, stream>>>(X, Ap, K, total);
    };
    auto gemm = [&](const _Float16* Ap, const _Float16* Bp, const float* bias,
                    float* O, int M, int Ktiles, int Npad, int Nvalid, int ldOut,
                    long long aS, long long bS, int batch,
                    int HHp, long long oSO, long long oSI, int mode) {
        dim3 grid((unsigned)(Npad / 64), (unsigned)(M / 64), (unsigned)batch);
        wmma_gemm_kernel<<<grid, 32, 0, stream>>>(Ap, Bp, bias, O, Ktiles, Nvalid,
                                                  ldOut, aS, bS, HHp, oSO, oSI, mode);
    };

    // ---- weight conversion (fp32 -> packed f16 fragments) ----
    for (int l = 0; l < GL; ++l) {
        packB(qkv_w  + (size_t)l * GC * (3*GC), qkvB + (size_t)l * GC * (3*GC), GC,   3*GC, 3*GC);
        packB(proj_w + (size_t)l * GC * GC,     prjB + (size_t)l * GC * GC,     GC,   GC,   GC);
        packB(fc_w   + (size_t)l * GC * (4*GC), fcB  + (size_t)l * GC * (4*GC), GC,   4*GC, 4*GC);
        packB(fc2_w  + (size_t)l * (4*GC) * GC, fc2B + (size_t)l * (4*GC) * GC, 4*GC, GC,   GC);
    }
    packB(lm_w, lmB, GC, GV, GVP);

    // ---- embedding ----
    {
        size_t total = (size_t)GBT * GC;
        embed_kernel<<<(unsigned)((total + 255) / 256), 256, 0, stream>>>(idx, wte, wpe, x);
    }

    const long long bhS   = (long long)GT * GD;        // 65536
    const long long scS   = (long long)GT * GT;        // 1048576
    const size_t qkvTot   = (size_t)GBT * 3 * GC;

    // ---- transformer blocks ----
    for (int l = 0; l < GL; ++l) {
        ln_pack_kernel<<<GBT, 256, 0, stream>>>(x, ln1_w + l*GC, ln1_b + l*GC, hA);
        gemm(hA, qkvB + (size_t)l*GC*(3*GC), qkv_b + l*(3*GC), qkv,
             GBT, GC/32, 3*GC, 3*GC, 3*GC, 0, 0, 1, 1, 0, 0, 0);

        qkv_pack_kernel<<<(unsigned)((qkvTot + 255) / 256), 256, 0, stream>>>(qkv, qA, kB, vB);

        // scores = (q/8) @ k^T   [per head: 1024 x 64 x 1024]
        gemm(qA, kB, nullptr, sc, GT, GD/32, GT, GT, GT,
             bhS, bhS, GB*GH, 1, scS, 0, 0);

        softmax_pack_kernel<<<GB*GH*GT, 256, 0, stream>>>(sc, Pp);

        // y = P @ v   [per head: 1024 x 1024 x 64], scatter into [B,T,C]
        gemm(Pp, vB, nullptr, y, GT, GT/32, GD, GD, GC,
             scS, bhS, GB*GH, GH, (long long)GT * GC, GD, 0);

        packA(y, yA, GBT, GC);
        gemm(yA, prjB + (size_t)l*GC*GC, proj_b + l*GC, x,
             GBT, GC/32, GC, GC, GC, 0, 0, 1, 1, 0, 0, 2);   // residual

        ln_pack_kernel<<<GBT, 256, 0, stream>>>(x, ln2_w + l*GC, ln2_b + l*GC, hA);
        gemm(hA, fcB + (size_t)l*GC*(4*GC), fc_b + l*(4*GC), h2,
             GBT, GC/32, 4*GC, 4*GC, 4*GC, 0, 0, 1, 1, 0, 0, 1);  // gelu

        packA(h2, h2A, GBT, 4*GC);
        gemm(h2A, fc2B + (size_t)l*(4*GC)*GC, fc2_b + l*GC, x,
             GBT, (4*GC)/32, GC, GC, GC, 0, 0, 1, 1, 0, 0, 2);   // residual
    }

    // ---- final LN + lm_head ----
    ln_pack_kernel<<<GBT, 256, 0, stream>>>(x, lnf_w, lnf_b, hA);
    gemm(hA, lmB, nullptr, logits, GBT, GC/32, GVP, GV, GV, 0, 0, 1, 1, 0, 0, 0);
}
